// Decoder_55551107007309
// MI455X (gfx1250) — matrix operand order repaired and checked by reading, not compiled, not run on hardware
//
#include <hip/hip_runtime.h>

typedef __attribute__((ext_vector_type(16))) __bf16 v16bf;
typedef __attribute__((ext_vector_type(8)))  float  v8f;

#define TT     64
#define TSTEPS 63
#define EE     256
#define DD     256

// ---------------- device helpers ----------------

__device__ inline float tanh_fast(float x) {
    x = fminf(15.f, fmaxf(-15.f, x));
    float e = __expf(2.f * x);
    return (e - 1.f) / (e + 1.f);
}
__device__ inline float sigmoid_fast(float x) {
    x = fminf(30.f, fmaxf(-30.f, x));
    return 1.f / (1.f + __expf(-x));
}
__device__ inline float wave_sum(float v) {
#pragma unroll
    for (int off = 16; off > 0; off >>= 1) v += __shfl_xor(v, off, 32);
    return v;
}
__device__ inline float wave_max(float v) {
#pragma unroll
    for (int off = 16; off > 0; off >>= 1) v = fmaxf(v, __shfl_xor(v, off, 32));
    return v;
}

// 16x32 bf16 A-matrix lane gather from row-major f32 (used once, in encproj):
// lanes 0-15: V0-3 -> K 0..7, V4-7 -> K 16..23 ; lanes 16-31: K 8..15 / 24..31
__device__ inline v16bf loadA_f32(const float* base, int stride, int k0, int lane) {
    int m = lane & 15, g = lane >> 4;
    const float* row = base + m * stride + k0;
    v16bf a;
#pragma unroll
    for (int p = 0; p < 8; ++p) {
        int kb = (p < 4) ? (8 * g + 2 * p) : (16 + 8 * g + 2 * (p - 4));
        a[2 * p]     = (__bf16)row[kb];
        a[2 * p + 1] = (__bf16)row[kb + 1];
    }
    return a;
}
// B tiles pre-packed: ((tile*8 + kstep)*32 + lane)*16 contiguous bf16 per lane
__device__ inline v16bf loadB_packed(const __bf16* pack, int tile, int ks, int lane) {
    return *(const v16bf*)(pack + ((((tile << 3) + ks) << 5) + lane) * 16);
}
__device__ inline v8f wmma_bf16(v16bf a, v16bf b, v8f c) {
    return __builtin_amdgcn_wmma_f32_16x16x32_bf16(false, a, false, b, (short)0, c, false, false);
}

// Scatter one (row m, col d) element into WMMA-A-layout LDS [8][32][16] bf16.
// Inverse of the A layout: d -> ks = d>>5, lane group g = (d>>3)&1,
// slot i = (d&7) | ((d>>4 & 1)<<3), lane = m + 16*g.
__device__ inline void storeA_pack(__bf16 (*pack)[32][16], int m, int d, float v) {
    int ks = d >> 5, kl = d & 31;
    int g  = (kl >> 3) & 1;
    int i  = (kl & 7) | (((kl >> 4) & 1) << 3);
    pack[ks][m + 16 * g][i] = (__bf16)v;
}
__device__ inline float loadA_pack_scalar(const __bf16 (*pack)[32][16], int m, int d) {
    int ks = d >> 5, kl = d & 31;
    int g  = (kl >> 3) & 1;
    int i  = (kl & 7) | (((kl >> 4) & 1) << 3);
    return (float)pack[ks][m + 16 * g][i];
}

// ---------------- kernel 1: weight packing into WMMA B-layout (bf16) ----------------
__global__ void pack_weights_kernel(const float* __restrict__ W_a1,
                                    const float* __restrict__ W_hh,
                                    unsigned short* __restrict__ wa1h_u,
                                    unsigned short* __restrict__ wa1c_u,
                                    unsigned short* __restrict__ wa1e_u,
                                    unsigned short* __restrict__ whh_u) {
    int idx = blockIdx.x * blockDim.x + threadIdx.x;
    const int SZ1 = 16 * 8 * 32 * 16;   // 65536 (256x256 weight)
    const int SZH = 64 * 8 * 32 * 16;   // 262144 (1024x256 weight)
    int which, local;
    if (idx < SZ1)            { which = 0; local = idx; }
    else if (idx < 2 * SZ1)   { which = 1; local = idx - SZ1; }
    else if (idx < 3 * SZ1)   { which = 2; local = idx - 2 * SZ1; }
    else if (idx < 3 * SZ1 + SZH) { which = 3; local = idx - 3 * SZ1; }
    else return;

    int i    = local & 15;
    int lane = (local >> 4) & 31;
    int ks   = (local >> 9) & 7;
    int tile = local >> 12;
    int k = ks * 32 + (((i >> 3) << 4) | ((lane >> 4) << 3) | (i & 7));
    int n = tile * 16 + (lane & 15);

    float v;
    unsigned short* dst;
    if      (which == 0) { v = W_a1[n * 768 + k];         dst = wa1h_u; } // hidden slice
    else if (which == 1) { v = W_a1[n * 768 + 256 + k];   dst = wa1c_u; } // cell slice
    else if (which == 2) { v = W_a1[n * 768 + 512 + k];   dst = wa1e_u; } // encoder slice
    else                 { v = W_hh[n * 256 + k];         dst = whh_u;  } // LSTM recurrent
    __bf16 bv = (__bf16)v;
    dst[local] = __builtin_bit_cast(unsigned short, bv);
}

// ---------------- kernel 2: enc_proj = input_encoded @ Wa1_enc.T + b_a1 ----------------
__global__ __launch_bounds__(1024) void encproj_kernel(const float* __restrict__ enc,
                                                       const unsigned short* __restrict__ wa1e_u,
                                                       const float* __restrict__ b_a1,
                                                       float* __restrict__ encp) {
    const __bf16* wa1e = (const __bf16*)wa1e_u;
    int gwid = (blockIdx.x * blockDim.x + threadIdx.x) >> 5;  // 8192 waves
    int lane = threadIdx.x & 31;
    int rt = gwid >> 4;   // 512 row tiles of [B*T, E]
    int ct = gwid & 15;   // 16 col tiles of E=256
    v8f acc = {};
#pragma unroll
    for (int ks = 0; ks < 8; ++ks) {
        v16bf a = loadA_f32(enc + (size_t)rt * 16 * EE, EE, ks * 32, lane);
        acc = wmma_bf16(a, loadB_packed(wa1e, ct, ks, lane), acc);
    }
    int n = ct * 16 + (lane & 15);
    int g = lane >> 4;
    float bias = b_a1[n];
#pragma unroll
    for (int r = 0; r < 8; ++r)
        encp[((size_t)rt * 16 + r + 8 * g) * EE + n] = acc[r] + bias;
}

// ---------------- kernel 3: persistent recurrence, one block per 16 batch rows ----------------
__global__ __launch_bounds__(512) void decoder_recur(
    const float* __restrict__ enc,        // [B,T,E]
    const float* __restrict__ yhist,      // [B,63]
    const float* __restrict__ encp,       // [B*T,E]
    const unsigned short* __restrict__ wa1h_u,
    const unsigned short* __restrict__ wa1c_u,
    const unsigned short* __restrict__ whh_u,
    const float* __restrict__ W_a2, const float* __restrict__ b_a2,
    const float* __restrict__ W_fc, const float* __restrict__ b_fc,
    const float* __restrict__ W_ih, const float* __restrict__ b_ih,
    const float* __restrict__ b_hh,
    const float* __restrict__ W_ff, const float* __restrict__ b_ff,
    float* __restrict__ out) {
    const __bf16* wa1h = (const __bf16*)wa1h_u;
    const __bf16* wa1c = (const __bf16*)wa1c_u;
    const __bf16* whh  = (const __bf16*)whh_u;

    // h and a bf16 shadow of c are kept directly in WMMA-A lane layout:
    // every A fetch is two ds_load_b128 per k-step instead of a 16-elem gather.
    __shared__ __bf16 sHp[8][32][16];   // 8KB  hidden, A-packed bf16
    __shared__ __bf16 sCp[8][32][16];   // 8KB  cell shadow, A-packed bf16
    __shared__ float  sC[16][256];      // 16KB cell state, f32 (cross-step accum)
    __shared__ float  sHC[16][256];     // 16KB hc_proj
    __shared__ __bf16 sCtx[16][256];    // 8KB  context
    __shared__ float  sScore[16][64];   // 4KB  scores -> attn (softmaxed in place)
    __shared__ float  sYt[16];

    const int tid  = threadIdx.x;
    const int w    = tid >> 5;      // wave id 0..15
    const int lane = tid & 31;
    const int b0   = blockIdx.x * 16;

    for (int i = tid; i < 16 * 256; i += 512) {
        (&sHp[0][0][0])[i] = (__bf16)0.f;
        (&sCp[0][0][0])[i] = (__bf16)0.f;
        (&sC[0][0])[i]     = 0.f;
    }
    __syncthreads();

    // per-lane constants for the score pass (e-range fixed per lane)
    float wa2r[8];
#pragma unroll
    for (int j = 0; j < 8; ++j) wa2r[j] = W_a2[lane * 8 + j];
    const float ba2 = b_a2[0];

    for (int s = 0; s < TSTEPS; ++s) {
        // ---- 1) hc_proj = h @ Wa1h.T + c @ Wa1c.T : wave w owns e-tile w ----
        {
            v8f acc = {};
#pragma unroll
            for (int ks = 0; ks < 8; ++ks) {
                v16bf a = *(const v16bf*)(&sHp[ks][lane][0]);
                acc = wmma_bf16(a, loadB_packed(wa1h, w, ks, lane), acc);
            }
#pragma unroll
            for (int ks = 0; ks < 8; ++ks) {
                v16bf a = *(const v16bf*)(&sCp[ks][lane][0]);
                acc = wmma_bf16(a, loadB_packed(wa1c, w, ks, lane), acc);
            }
            int n = w * 16 + (lane & 15);
            int g = lane >> 4;
#pragma unroll
            for (int r = 0; r < 8; ++r) sHC[r + 8 * g][n] = acc[r];
        }
        __syncthreads();

        // ---- 2) scores + softmax : wave w owns batch row b=w ----
        {
            const int b = w;
            float hcr[8];
#pragma unroll
            for (int j = 0; j < 8; ++j) hcr[j] = sHC[b][lane * 8 + j];
            const float* ep0 = encp + ((size_t)(b0 + b) * TT) * EE + lane * 8;
            for (int t = 0; t < TT; ++t) {
                const float* ep = ep0 + t * EE;
                if (t + 1 < TT) __builtin_prefetch(ep + EE, 0, 3);  // global_prefetch_b8
                float p = 0.f;
#pragma unroll
                for (int j = 0; j < 8; ++j) p += wa2r[j] * tanh_fast(hcr[j] + ep[j]);
                p = wave_sum(p);
                if (lane == 0) sScore[b][t] = p + ba2;
            }
            // softmax over T=64 within the same wave (LDS per-wave in-order)
            float s0 = sScore[b][lane], s1 = sScore[b][lane + 32];
            float mx = wave_max(fmaxf(s0, s1));
            float e0 = __expf(s0 - mx), e1 = __expf(s1 - mx);
            float inv = 1.f / wave_sum(e0 + e1);
            sScore[b][lane]      = e0 * inv;
            sScore[b][lane + 32] = e1 * inv;
        }
        __syncthreads();

        // ---- 3) context[b][e] = sum_t attn[b][t] * enc[b,t,e] ----
#pragma unroll
        for (int r = 0; r < 8; ++r) {
            int idx = tid + r * 512;
            int b = idx >> 8, e = idx & 255;
            const float* ebase = enc + ((size_t)(b0 + b) * TT) * EE + e;
            float a0 = 0.f;
            for (int t = 0; t < TT; ++t) {
                if (t + 1 < TT) __builtin_prefetch(ebase + (t + 1) * EE, 0, 3);
                a0 += sScore[b][t] * ebase[t * EE];
            }
            sCtx[b][e] = (__bf16)a0;
        }
        __syncthreads();

        // ---- 4) y_tilde : wave w owns b=w ----
        {
            const int b = w;
            float p = 0.f;
#pragma unroll
            for (int j = 0; j < 8; ++j) {
                int e = lane * 8 + j;
                p += W_fc[e] * (float)sCtx[b][e];
            }
            p = wave_sum(p);
            if (lane == 0)
                sYt[b] = p + W_fc[EE] * yhist[(b0 + b) * TSTEPS + s] + b_fc[0];
        }
        __syncthreads();

        // ---- 5) gates = h @ W_hh.T (+ y_tilde*W_ih + biases), fused LSTM update ----
        // wave w computes gate tiles {w, 16+w, 32+w, 48+w} -> i/f/g/o for
        // d in [16w, 16w+16) stay in this wave's registers.
        {
            v8f gi = {}, gf = {}, gg = {}, go = {};
#pragma unroll
            for (int ks = 0; ks < 8; ++ks) {
                v16bf a = *(const v16bf*)(&sHp[ks][lane][0]);
                gi = wmma_bf16(a, loadB_packed(whh,      w, ks, lane), gi);
                gf = wmma_bf16(a, loadB_packed(whh, 16 + w, ks, lane), gf);
                gg = wmma_bf16(a, loadB_packed(whh, 32 + w, ks, lane), gg);
                go = wmma_bf16(a, loadB_packed(whh, 48 + w, ks, lane), go);
            }
            int d = w * 16 + (lane & 15);
            int g = lane >> 4;
            float wih_i = W_ih[d],          bi = b_ih[d]          + b_hh[d];
            float wih_f = W_ih[DD + d],     bf = b_ih[DD + d]     + b_hh[DD + d];
            float wih_g = W_ih[2 * DD + d], bg = b_ih[2 * DD + d] + b_hh[2 * DD + d];
            float wih_o = W_ih[3 * DD + d], bo = b_ih[3 * DD + d] + b_hh[3 * DD + d];
            __syncthreads();   // all waves finished reading sHp as A before overwrite
#pragma unroll
            for (int r = 0; r < 8; ++r) {
                int m = r + 8 * g;
                float yt = sYt[m];
                float iv = sigmoid_fast(gi[r] + yt * wih_i + bi);
                float fv = sigmoid_fast(gf[r] + yt * wih_f + bf);
                float gv = tanh_fast   (gg[r] + yt * wih_g + bg);
                float ov = sigmoid_fast(go[r] + yt * wih_o + bo);
                float cn = fv * sC[m][d] + iv * gv;
                sC[m][d] = cn;
                storeA_pack(sCp, m, d, cn);
                storeA_pack(sHp, m, d, ov * tanh_fast(cn));
            }
        }
        __syncthreads();
    }

    // ---- final: out[b, 0:2] = concat(h, context) @ W_ff.T + b_ff ----
    {
        const int b = w;
#pragma unroll
        for (int k = 0; k < 2; ++k) {
            float p = 0.f;
#pragma unroll
            for (int j = 0; j < 16; ++j) {
                int idx = lane * 16 + j;
                float v = (idx < DD) ? loadA_pack_scalar(sHp, b, idx)
                                     : (float)sCtx[b][idx - DD];
                p += W_ff[k * (DD + EE) + idx] * v;
            }
            p = wave_sum(p);
            if (lane == 0) out[(b0 + b) * 2 + k] = p + b_ff[k];
        }
    }
}

// ---------------- host launcher ----------------
extern "C" void kernel_launch(void* const* d_in, const int* in_sizes, int n_in,
                              void* d_out, int out_size, void* d_ws, size_t ws_size,
                              hipStream_t stream) {
    (void)in_sizes; (void)n_in; (void)out_size; (void)ws_size;
    const float* enc   = (const float*)d_in[0];
    const float* yhist = (const float*)d_in[1];
    const float* W_a1  = (const float*)d_in[2];
    const float* b_a1  = (const float*)d_in[3];
    const float* W_a2  = (const float*)d_in[4];
    const float* b_a2  = (const float*)d_in[5];
    const float* W_fc  = (const float*)d_in[6];
    const float* b_fc  = (const float*)d_in[7];
    const float* W_ih  = (const float*)d_in[8];
    const float* W_hh  = (const float*)d_in[9];
    const float* b_ih  = (const float*)d_in[10];
    const float* b_hh  = (const float*)d_in[11];
    const float* W_ff  = (const float*)d_in[12];
    const float* b_ff  = (const float*)d_in[13];
    float* out = (float*)d_out;

    // workspace layout: enc_proj (8 MB f32) then 4 packed bf16 weight blobs
    char* ws = (char*)d_ws;
    float* encp = (float*)ws;
    unsigned short* wa1h = (unsigned short*)(ws + (size_t)8 * 1024 * 1024);
    unsigned short* wa1c = wa1h + 65536;
    unsigned short* wa1e = wa1c + 65536;
    unsigned short* whh  = wa1e + 65536;

    // total pack elements: 3*65536 + 262144 = 458752 -> 1792 blocks of 256
    pack_weights_kernel<<<1792, 256, 0, stream>>>(W_a1, W_hh, wa1h, wa1c, wa1e, whh);
    // 8192 waves = 256 blocks x 1024 threads
    encproj_kernel<<<256, 1024, 0, stream>>>(enc, wa1e, b_a1, encp);
    // 8 persistent blocks, 16 waves each, one block per 16 batch rows
    decoder_recur<<<8, 512, 0, stream>>>(enc, yhist, encp, wa1h, wa1c, whh,
                                         W_a2, b_a2, W_fc, b_fc,
                                         W_ih, b_ih, b_hh, W_ff, b_ff, out);
}
